// self_attn_11330123727100
// MI455X (gfx1250) — matrix operand order repaired and checked by reading, not compiled, not run on hardware
//
#include <hip/hip_runtime.h>
#include <hip/hip_bf16.h>

// ---------------------------------------------------------------------------
// Fused sigmoid-attention for MI455X (gfx1250, wave32, WMMA).
//   y = gamma * (V . sigmoid(Q^T K)^T) + x,  B=2, C=64, N=96*96=9216, C8=8
// Flash-style fusion, f16 WMMA 16x16x32 with f32 accumulation.
// K/V tiles are staged into LDS once per block via async-to-LDS copies
// (double buffered, ASYNCcnt, inline asm) so the 8 waves of a block share
// them instead of re-streaming from L2. j-split + deterministic reduction.
// ---------------------------------------------------------------------------

#define BATCH  2
#define CCH    64
#define HW     9216          // 96*96 tokens
#define ITILES (HW / 16)     // 576 query tiles per batch
#define NTASK  (BATCH * ITILES)
#define JSTAGE 128           // tokens staged to LDS per double-buffer stage
#define KPAD   40            // padded K row (halves)  : 80B  = 5*16
#define VPAD   136           // padded V row (halves)  : 272B = 17*16
#define PPAD   40            // padded P row (halves)

typedef __attribute__((ext_vector_type(16))) _Float16 v16h;
typedef __attribute__((ext_vector_type(8)))  _Float16 v8h;
typedef __attribute__((ext_vector_type(8)))  float    v8f;
typedef __attribute__((ext_vector_type(4)))  float    v4f;

union HV16 { v16h v; v8h h[2]; };

// Async 16B global->LDS copy (ASYNCcnt-tracked).  ISA §15.18.3 opcode 98.
// VDST holds the per-lane LDS byte offset; AS3 pointers are 32-bit offsets.
__device__ __forceinline__ void async_cp16(const _Float16* g, _Float16* l) {
  const unsigned ldsoff =
      (unsigned)(size_t)(__attribute__((address_space(3))) _Float16*)l;
  asm volatile("global_load_async_to_lds_b128 %0, %1, off"
               :: "v"(ldsoff), "v"(g) : "memory");
}

// ---------------------------------------------------------------------------
// Kernel 1: projections.
//   Qt[b][n][16]  f16, channels 8..15 zero          (A-fragment feed)
//   Kt[b][n][32]  f16, channels 8..31 zero          (B-fragment feed)
//   Vh[b][c][n]   f16 row-major                     (A-fragment feed)
// ---------------------------------------------------------------------------
__global__ __launch_bounds__(256) void sattn_proj_kernel(
    const float* __restrict__ x,
    const float* __restrict__ Wq, const float* __restrict__ bq,
    const float* __restrict__ Wk, const float* __restrict__ bk,
    const float* __restrict__ Wv, const float* __restrict__ bv,
    _Float16* __restrict__ Qt, _Float16* __restrict__ Kt,
    _Float16* __restrict__ Vh)
{
  __shared__ float xs[64][65];
  const int tid = threadIdx.x;
  const int b   = blockIdx.x / (HW / 64);
  const int n0  = (blockIdx.x % (HW / 64)) * 64;

  for (int idx = tid; idx < 64 * 64; idx += 256) {
    const int c = idx >> 6, t = idx & 63;
    xs[c][t] = x[((size_t)(b * CCH + c)) * HW + n0 + t];
  }
  __syncthreads();

  // 112 output rows: 0..15 -> Qt (8..15 pad), 16..47 -> Kt (24..47 pad),
  // 48..111 -> Vh.  `o` is wave-uniform -> scalar weight loads.
  for (int oidx = tid; oidx < 112 * 64; oidx += 256) {
    const int o = oidx >> 6;
    const int t = oidx & 63;
    const int n = n0 + t;
    if (o < 16) {
      _Float16 r = (_Float16)0.f;
      if (o < 8) {
        float s = bq[o];
#pragma unroll
        for (int cc = 0; cc < 64; ++cc) s += Wq[o * 64 + cc] * xs[cc][t];
        r = (_Float16)s;
      }
      Qt[((size_t)b * HW + n) * 16 + o] = r;
    } else if (o < 48) {
      const int oo = o - 16;
      _Float16 r = (_Float16)0.f;
      if (oo < 8) {
        float s = bk[oo];
#pragma unroll
        for (int cc = 0; cc < 64; ++cc) s += Wk[oo * 64 + cc] * xs[cc][t];
        r = (_Float16)s;
      }
      Kt[((size_t)b * HW + n) * 32 + oo] = r;
    } else {
      const int c = o - 48;
      float s = bv[c];
#pragma unroll
      for (int cc = 0; cc < 64; ++cc) s += Wv[c * 64 + cc] * xs[cc][t];
      Vh[((size_t)(b * CCH + c)) * HW + n] = (_Float16)s;
    }
  }
}

// ---------------------------------------------------------------------------
// Stage one JSTAGE-token slab of K (8KB) + V (16KB) into LDS.
// 1536 x 16B chunks over 256 threads = 6 async instructions per wave.
// ---------------------------------------------------------------------------
__device__ __forceinline__ void stage_copy(
    _Float16* __restrict__ kl, _Float16* __restrict__ vl,
    const _Float16* __restrict__ kg, const _Float16* __restrict__ vg,
    int jstage, int tid)
{
#pragma unroll
  for (int it = 0; it < 6; ++it) {
    const int c = tid + it * 256;
    if (c < 512) {                                    // K: rows of 64B
      const int row = c >> 2, seg = c & 3;
      async_cp16(kg + ((size_t)(jstage + row)) * 32 + seg * 8,
                 kl + row * KPAD + seg * 8);
    } else {                                          // V: 64 rows x 256B
      const int cc = c - 512;
      const int row = cc >> 4, seg = cc & 15;
      async_cp16(vg + (size_t)row * HW + jstage + seg * 8,
                 vl + row * VPAD + seg * 8);
    }
  }
}

// ---------------------------------------------------------------------------
// Kernel 2: fused attention.  One wave = one 16-query tile x all 64 output
// channels over a 1/jsplit slice of the j range; 8 waves per block share
// LDS-staged K/V slabs.
// ---------------------------------------------------------------------------
__global__ __launch_bounds__(256) void sattn_fused_kernel(
    const float* __restrict__ x,
    const _Float16* __restrict__ Qt, const _Float16* __restrict__ Kt,
    const _Float16* __restrict__ Vh, const float* __restrict__ gamma,
    float* __restrict__ out, float* __restrict__ part, int jsplit)
{
  __shared__ _Float16 Kbuf[2][JSTAGE][KPAD];   // 20.0 KB
  __shared__ _Float16 Vbuf[2][CCH][VPAD];      // 34.0 KB
  __shared__ _Float16 Plds[8][16][PPAD];       // 10.0 KB  (total = 64 KB)

  const int tid  = threadIdx.x;
  const int wv   = tid >> 5;
  const int lane = tid & 31;
  const int half = lane >> 4;
  const int l15  = lane & 15;

  const int task   = blockIdx.x * 8 + wv;      // 0 .. NTASK*jsplit-1
  const int itile  = task % NTASK;
  const int js     = task / NTASK;
  const int b      = itile / ITILES;
  const int ibase  = (itile % ITILES) * 16;
  const int jchunk = HW / jsplit;
  const int jbeg   = js * jchunk;

  // --- A fragment of Q (16x32): lane(l15)=row M; VGPR0-3 = K[half*8..+7],
  //     VGPR4-7 = K[16+half*8..] (zero: only 8 real channels). ---
  HV16 aq;
  {
    const _Float16* qrow =
        Qt + ((size_t)b * HW + (ibase + l15)) * 16 + (size_t)half * 8;
    aq.h[0] = *(const v8h*)qrow;
    v8h z;
#pragma unroll
    for (int e = 0; e < 8; ++e) z[e] = (_Float16)0.f;
    aq.h[1] = z;
  }

  v8f acc[4];
#pragma unroll
  for (int ct = 0; ct < 4; ++ct) { v8f z = {}; acc[ct] = z; }

  const _Float16* kb = Kt + (size_t)b * HW * 32;
  const _Float16* vb = Vh + (size_t)b * CCH * HW;

  const int nstages = jchunk / JSTAGE;
  stage_copy(&Kbuf[0][0][0], &Vbuf[0][0][0], kb, vb, jbeg, tid);

  for (int s = 0; s < nstages; ++s) {
    const int cur = s & 1;
    if (s + 1 < nstages) {
      stage_copy(&Kbuf[cur ^ 1][0][0], &Vbuf[cur ^ 1][0][0], kb, vb,
                 jbeg + (s + 1) * JSTAGE, tid);
      asm volatile("s_wait_asynccnt 6" ::: "memory");  // cur stage landed
    } else {
      asm volatile("s_wait_asynccnt 0" ::: "memory");
    }
    __syncthreads();

    // Fully unrolled: 4 independent 32-token bodies (24 WMMAs) so the
    // scheduler can pipeline LDS loads over WMMA/sigmoid chains.
#pragma unroll
    for (int jo = 0; jo < JSTAGE; jo += 32) {
      // ---- S = Q^T K, two 16-wide j sub-tiles, sigmoid, stage as f16 ----
#pragma unroll
      for (int sub = 0; sub < 2; ++sub) {
        HV16 bk;
        const _Float16* kr = &Kbuf[cur][jo + sub * 16 + l15][half * 16];
        bk.h[0] = *(const v8h*)kr;
        bk.h[1] = *(const v8h*)(kr + 8);
        v8f zc = {};
        v8f sres = __builtin_amdgcn_wmma_f32_16x16x32_f16(
            false, aq.v, false, bk.v, (short)0, zc, false, false);
        // D layout: VGPR r -> row i = r + 8*half, col j = l15
#pragma unroll
        for (int r = 0; r < 8; ++r) {
          const float p = __builtin_amdgcn_rcpf(
              1.0f + __builtin_amdgcn_exp2f(-1.4426950408889634f * sres[r]));
          Plds[wv][r + 8 * half][sub * 16 + l15] = (_Float16)p;
        }
      }
      __builtin_amdgcn_wave_barrier();         // order LDS stores vs. loads

      // ---- B fragment of P^T: lane reads P[i][half*16..+15] -------------
      HV16 bp;
      {
        const _Float16* pr = &Plds[wv][l15][half * 16];
        bp.h[0] = *(const v8h*)pr;
        bp.h[1] = *(const v8h*)(pr + 8);
      }

      // ---- Out += V . P^T : 4 channel tiles -----------------------------
#pragma unroll
      for (int ct = 0; ct < 4; ++ct) {
        const _Float16* vr = &Vbuf[cur][ct * 16 + l15][jo + half * 16];
        HV16 av;
        av.h[0] = *(const v8h*)vr;
        av.h[1] = *(const v8h*)(vr + 8);
        acc[ct] = __builtin_amdgcn_wmma_f32_16x16x32_f16(
            false, av.v, false, bp.v, (short)0, acc[ct], false, false);
      }
    }
    __syncthreads();                           // all waves done with `cur`
  }

  // ---- epilogue --------------------------------------------------------
  const size_t base = (size_t)b * CCH * HW;
  if (jsplit == 1) {
    const float g = gamma[0];
#pragma unroll
    for (int ct = 0; ct < 4; ++ct) {
#pragma unroll
      for (int r = 0; r < 8; ++r) {
        const int c = ct * 16 + r + 8 * half;
        const size_t idx = base + (size_t)c * HW + ibase + l15;
        out[idx] = g * acc[ct][r] + x[idx];
      }
    }
  } else {
    float* pb = part + (size_t)js * (BATCH * CCH * HW);
#pragma unroll
    for (int ct = 0; ct < 4; ++ct) {
#pragma unroll
      for (int r = 0; r < 8; ++r) {
        const int c = ct * 16 + r + 8 * half;
        pb[base + (size_t)c * HW + ibase + l15] = acc[ct][r];
      }
    }
  }
}

// ---------------------------------------------------------------------------
// Kernel 3: deterministic reduction of j-split partials: y = g*sum + x.
// ---------------------------------------------------------------------------
__global__ __launch_bounds__(256) void sattn_reduce_kernel(
    const float* __restrict__ x, const float* __restrict__ part,
    const float* __restrict__ gamma, float* __restrict__ out, int jsplit)
{
  const size_t i = (size_t)blockIdx.x * 256 + threadIdx.x;   // float4 index
  const size_t stride4 = (size_t)(BATCH * CCH * HW) / 4;
  const float g = gamma[0];
  const v4f* p4 = (const v4f*)part;
  v4f s = p4[i];
  for (int js = 1; js < jsplit; ++js) {
    const v4f t = p4[i + (size_t)js * stride4];
    s.x += t.x; s.y += t.y; s.z += t.z; s.w += t.w;
  }
  const v4f xv = ((const v4f*)x)[i];
  v4f o;
  o.x = g * s.x + xv.x; o.y = g * s.y + xv.y;
  o.z = g * s.z + xv.z; o.w = g * s.w + xv.w;
  ((v4f*)out)[i] = o;
}

// ---------------------------------------------------------------------------
extern "C" void kernel_launch(void* const* d_in, const int* in_sizes, int n_in,
                              void* d_out, int out_size, void* d_ws,
                              size_t ws_size, hipStream_t stream) {
  (void)in_sizes; (void)n_in; (void)out_size;
  const float* x     = (const float*)d_in[0];
  const float* Wq    = (const float*)d_in[1];
  const float* bq    = (const float*)d_in[2];
  const float* Wk    = (const float*)d_in[3];
  const float* bk    = (const float*)d_in[4];
  const float* Wv    = (const float*)d_in[5];
  const float* bv    = (const float*)d_in[6];
  const float* gamma = (const float*)d_in[7];
  float* out = (float*)d_out;

  // Workspace: Qt | Kt | Vh (f16) | partials (f32)
  _Float16* Qt = (_Float16*)d_ws;
  _Float16* Kt = Qt + (size_t)BATCH * HW * 16;
  _Float16* Vh = Kt + (size_t)BATCH * HW * 32;
  float*  part = (float*)(Vh + (size_t)BATCH * CCH * HW);
  const size_t projBytes = (size_t)BATCH * HW * (16 + 32 + CCH) * sizeof(_Float16);
  const size_t sliceBytes = (size_t)BATCH * CCH * HW * sizeof(float);

  // Largest j-split whose partial buffers fit (ws_size is run-constant, so
  // this choice — and the summation order — is deterministic).
  int jsplit = 1;
  if (ws_size >= projBytes + 4 * sliceBytes)      jsplit = 4;
  else if (ws_size >= projBytes + 2 * sliceBytes) jsplit = 2;

  sattn_proj_kernel<<<BATCH * (HW / 64), 256, 0, stream>>>(
      x, Wq, bq, Wk, bk, Wv, bv, Qt, Kt, Vh);
  sattn_fused_kernel<<<(NTASK * jsplit) / 8, 256, 0, stream>>>(
      x, Qt, Kt, Vh, gamma, out, part, jsplit);
  if (jsplit > 1) {
    sattn_reduce_kernel<<<(BATCH * CCH * HW) / (4 * 256), 256, 0, stream>>>(
        x, part, gamma, out, jsplit);
  }
}
